// MultiHeadAttention_249108103258
// MI455X (gfx1250) — matrix operand order repaired and checked by reading, not compiled, not run on hardware
//
#include <hip/hip_runtime.h>
#include <math.h>

// MI455X / gfx1250 wave32 WMMA multi-head attention.
// GEMMs via v_wmma_f32_16x16x32_f16; attention K/V staged through LDS with
// gfx1250 async global->LDS copies (ASYNCcnt), double-buffered.

typedef __attribute__((ext_vector_type(16))) _Float16 v16h;
typedef __attribute__((ext_vector_type(8)))  _Float16 v8h;
typedef __attribute__((ext_vector_type(4)))  _Float16 v4h;
typedef __attribute__((ext_vector_type(8)))  float    v8f;
typedef __attribute__((ext_vector_type(4)))  float    v4f;
typedef __attribute__((ext_vector_type(4)))  int      v4i;

#define Bc  4
#define Sc  2048
#define Ec  1024
#define Hc  16
#define DHc 64

// LDS row strides (in halves) chosen for conflict-free b128 reads:
// K rows: 72 halves = 144B (36 dw: 36*r mod 64 distinct for r=0..15)
// V rows: 40 halves =  80B (20 dw: 20*r mod 64 distinct for r=0..15)
#define KROW 72
#define VROW 40

static __device__ __forceinline__ v8f wmma16(v16h a, v16h b, v8f c) {
  return __builtin_amdgcn_wmma_f32_16x16x32_f16(false, a, false, b, (short)0, c, false, false);
}

// A-matrix fragment (16x32 f16, MxK): lane m=l%16, halves i -> K=16*hs+i.
static __device__ __forceinline__ v16h load_afrag(const _Float16* rowp, int hs) {
  v8h lo = *(const v8h*)(rowp + 16 * hs);
  v8h hi = *(const v8h*)(rowp + 16 * hs + 8);
  v16h f;
#pragma unroll
  for (int i = 0; i < 8; ++i) { f[i] = lo[i]; f[i + 8] = hi[i]; }
  return f;
}

// B-matrix fragment (32x16 f16, KxN): lane n=l%16, halves i -> K=16*hs+i.
static __device__ __forceinline__ v16h load_bfrag(const _Float16* rowp, int hs) {
  v8h lo = *(const v8h*)(rowp + 16 * hs);
  v8h hi = *(const v8h*)(rowp + 16 * hs + 8);
  v16h f;
#pragma unroll
  for (int i = 0; i < 8; ++i) { f[i] = lo[i]; f[i + 8] = hi[i]; }
  return f;
}

// ---- gfx1250 async global -> LDS copy (16B per lane), ASYNCcnt-tracked ----
// Builtin signature (from hipcc diagnostic): (int4 AS(1)*, int4 AS(3)*, Imm, Imm).
static __device__ __forceinline__ void async_ld_b128(const _Float16* g, _Float16* s) {
#if __has_builtin(__builtin_amdgcn_global_load_async_to_lds_b128)
  __builtin_amdgcn_global_load_async_to_lds_b128(
      (__attribute__((address_space(1))) v4i*)(unsigned long long)(uintptr_t)g,
      (__attribute__((address_space(3))) v4i*)(unsigned)(uintptr_t)s,
      0, 0);
#else
  unsigned loff = (unsigned)(uintptr_t)s;  // low 32 bits of generic addr = LDS offset
  asm volatile("global_load_async_to_lds_b128 %0, %1, off"
               :: "v"(loff), "v"((unsigned long long)(uintptr_t)g)
               : "memory");
#endif
}

static __device__ __forceinline__ void wait_async_le2() {
#if __has_builtin(__builtin_amdgcn_s_wait_asynccnt)
  __builtin_amdgcn_s_wait_asynccnt(2);
#else
  asm volatile("s_wait_asynccnt 0x2" ::: "memory");
#endif
}
static __device__ __forceinline__ void wait_async_le0() {
#if __has_builtin(__builtin_amdgcn_s_wait_asynccnt)
  __builtin_amdgcn_s_wait_asynccnt(0);
#else
  asm volatile("s_wait_asynccnt 0x0" ::: "memory");
#endif
}

// ---------------- conversion kernels ----------------
__global__ void k_cvt_f2h(const float* __restrict__ src, _Float16* __restrict__ dst, int n) {
  int i = (blockIdx.x * blockDim.x + threadIdx.x) * 4;
  if (i + 3 < n) {
    float4 v = *(const float4*)(src + i);
    v4h o;
    o[0] = (_Float16)v.x; o[1] = (_Float16)v.y; o[2] = (_Float16)v.z; o[3] = (_Float16)v.w;
    *(v4h*)(dst + i) = o;
  }
}

// W [H,E,DH] (f32) -> Wt [H,DH,E] (f16)
__global__ void k_cvt_wT(const float* __restrict__ W, _Float16* __restrict__ Wt) {
  int idx = blockIdx.x * blockDim.x + threadIdx.x;  // over H*DH*E
  int h = idx / (DHc * Ec);
  int r = idx % (DHc * Ec);
  int d = r / Ec;
  int e = r % Ec;
  Wt[idx] = (_Float16)W[((size_t)h * Ec + e) * DHc + d];
}

// ---------------- QKV projection ----------------
__global__ void k_qkv(const _Float16* __restrict__ xh,
                      const _Float16* __restrict__ wqt, const _Float16* __restrict__ wkt,
                      const _Float16* __restrict__ wvt,
                      const float* __restrict__ bq, const float* __restrict__ bk,
                      const float* __restrict__ bv,
                      _Float16* __restrict__ qh, _Float16* __restrict__ kh,
                      _Float16* __restrict__ vtp) {
  const int tid  = blockIdx.x * 256 + threadIdx.x;
  const int wid  = tid >> 5;
  const int lane = threadIdx.x & 31;
  const int lq   = lane & 15;
  const int hs   = lane >> 4;

  const int mat = wid / (Bc * Hc * (Sc / 16));
  int r         = wid % (Bc * Hc * (Sc / 16));
  const int bh  = r / (Sc / 16);
  const int st  = r % (Sc / 16);
  const int b   = bh / Hc;
  const int h   = bh % Hc;

  const _Float16* xrow = xh + ((size_t)(b * Sc + st * 16 + lq)) * Ec;

  if (mat < 2) {
    const _Float16* wt = (mat ? wkt : wqt) + (size_t)h * DHc * Ec;
    const float* bias  = (mat ? bk : bq) + h * DHc;
    _Float16* outp     = (mat ? kh : qh) + ((size_t)bh * Sc + st * 16) * DHc;
    v8f acc[4] = {};
    for (int e = 0; e < Ec; e += 32) {
      v16h bfr = load_bfrag(xrow + e, hs);
#pragma unroll
      for (int t = 0; t < 4; ++t) {
        v16h afr = load_afrag(wt + (size_t)(16 * t + lq) * Ec + e, hs);
        acc[t] = wmma16(afr, bfr, acc[t]);
      }
    }
#pragma unroll
    for (int t = 0; t < 4; ++t) {
      const float* bp = bias + 16 * t + 8 * hs;
      v8h pk;
#pragma unroll
      for (int j = 0; j < 8; ++j) pk[j] = (_Float16)(acc[t][j] + bp[j]);
      *(v8h*)(outp + (size_t)lq * DHc + 16 * t + 8 * hs) = pk;
    }
  } else {
    const _Float16* wt = wvt + (size_t)h * DHc * Ec;
    const float* bias  = bv + h * DHc;
    _Float16* outp     = vtp + (size_t)bh * DHc * Sc + st * 16;
    v8f acc[4] = {};
    for (int e = 0; e < Ec; e += 32) {
      v16h afr = load_afrag(xrow + e, hs);
#pragma unroll
      for (int t = 0; t < 4; ++t) {
        v16h bfr = load_bfrag(wt + (size_t)(16 * t + lq) * Ec + e, hs);
        acc[t] = wmma16(afr, bfr, acc[t]);
      }
    }
#pragma unroll
    for (int t = 0; t < 4; ++t) {
      float bb = bias[16 * t + lq];
      v8h pk;
#pragma unroll
      for (int j = 0; j < 8; ++j) pk[j] = (_Float16)(acc[t][j] + bb);
      *(v8h*)(outp + (size_t)(16 * t + lq) * Sc + 8 * hs) = pk;
    }
  }
}

// ---------------- flash attention (transposed, LDS-staged K/V) ----------------
// Block = 8 waves, all sharing (b,h); each wave owns 16 queries. K/V chunks of
// 32 keys are async-copied to LDS once per block (8x traffic reduction) and
// double-buffered; online softmax state is a per-lane scalar.
__global__ void k_attn(const _Float16* __restrict__ qh, const _Float16* __restrict__ kh,
                       const _Float16* __restrict__ vtp, _Float16* __restrict__ ctx) {
  __shared__ __align__(16) _Float16 ksm[2][32 * KROW];
  __shared__ __align__(16) _Float16 vsm[2][64 * VROW];

  const int tix  = threadIdx.x;
  const int tid  = blockIdx.x * 256 + tix;
  const int wid  = tid >> 5;
  const int lane = tix & 31;
  const int lq   = lane & 15;
  const int hs   = lane >> 4;

  const int bh = wid / (Sc / 16);
  const int st = wid % (Sc / 16);
  const int b  = bh / Hc;
  const int h  = bh % Hc;

  const _Float16* qrow = qh + ((size_t)bh * Sc + st * 16 + lq) * DHc;
  v16h qb0 = load_bfrag(qrow, hs);
  v16h qb1 = load_bfrag(qrow + 32, hs);

  // async copy assignments: 256 threads x one 16B segment per tile
  const int krow = tix >> 3, kseg = tix & 7;  // K: 32 rows x (64 halves = 8 segs)
  const int vrow = tix >> 2, vseg = tix & 3;  // V: 64 rows x (32 halves = 4 segs)
  const _Float16* kg = kh + (size_t)bh * Sc * DHc + (size_t)krow * DHc + kseg * 8;
  const _Float16* vg = vtp + ((size_t)bh * DHc + vrow) * Sc + vseg * 8;
  _Float16* ksd[2] = { &ksm[0][krow * KROW + kseg * 8], &ksm[1][krow * KROW + kseg * 8] };
  _Float16* vsd[2] = { &vsm[0][vrow * VROW + vseg * 8], &vsm[1][vrow * VROW + vseg * 8] };

  const float scale = 0.03125f;  // 1/sqrt(E)
  float mrow = -1e30f, lrow = 0.f;
  v8f acc[4] = {};

  // prologue: stage chunk 0 into buffer 0
  async_ld_b128(kg, ksd[0]);
  async_ld_b128(vg, vsd[0]);

  for (int i = 0; i < Sc / 32; ++i) {
    const int cur = i & 1;
    const int k0  = i * 32;
    if (i + 1 < Sc / 32) {
      async_ld_b128(kg + (size_t)(k0 + 32) * DHc, ksd[cur ^ 1]);
      async_ld_b128(vg + (k0 + 32),               vsd[cur ^ 1]);
      wait_async_le2();   // chunk i complete (in-order), chunk i+1 in flight
    } else {
      wait_async_le0();
    }
    __syncthreads();

    const _Float16* kr0 = &ksm[cur][(size_t)lq * KROW];
    const _Float16* kr1 = &ksm[cur][(size_t)(16 + lq) * KROW];
    v16h ka00 = load_afrag(kr0, hs), ka01 = load_afrag(kr0 + 32, hs);
    v16h ka10 = load_afrag(kr1, hs), ka11 = load_afrag(kr1 + 32, hs);

    v8f z = {};
    v8f s0 = wmma16(ka00, qb0, z); s0 = wmma16(ka01, qb1, s0);  // keys k0..k0+15
    v8f s1 = wmma16(ka10, qb0, z); s1 = wmma16(ka11, qb1, s1);  // keys k0+16..k0+31

    float p0[8], p1[8];
    float cm = -1e30f;
#pragma unroll
    for (int j = 0; j < 8; ++j) {
      p0[j] = s0[j] * scale;
      p1[j] = s1[j] * scale;
      cm = fmaxf(cm, fmaxf(p0[j], p1[j]));
    }
    cm = fmaxf(cm, __shfl_xor(cm, 16, 32));
    float mnew  = fmaxf(mrow, cm);
    float alpha = __expf(mrow - mnew);
    float ls = 0.f;
#pragma unroll
    for (int j = 0; j < 8; ++j) {
      p0[j] = __expf(p0[j] - mnew);
      p1[j] = __expf(p1[j] - mnew);
      ls += p0[j] + p1[j];
    }
    ls += __shfl_xor(ls, 16, 32);
    lrow = lrow * alpha + ls;
    mrow = mnew;
#pragma unroll
    for (int t = 0; t < 4; ++t)
#pragma unroll
      for (int j = 0; j < 8; ++j) acc[t][j] *= alpha;

    // Repack P^T (two C-tiles) into B-fragment: lane needs keys 16hs..16hs+15.
    v16h pb;
#pragma unroll
    for (int j = 0; j < 8; ++j) {
      float send = hs ? p0[j] : p1[j];
      float ex   = __shfl_xor(send, 16, 32);
      float lo   = hs ? ex : p0[j];
      float hi   = hs ? p1[j] : ex;
      pb[j]     = (_Float16)lo;
      pb[j + 8] = (_Float16)hi;
    }

#pragma unroll
    for (int t = 0; t < 4; ++t) {
      const _Float16* vr = &vsm[cur][(size_t)(16 * t + lq) * VROW];
      acc[t] = wmma16(load_afrag(vr, hs), pb, acc[t]);
    }
    __syncthreads();  // all waves done with buffer `cur` before it is refilled
  }

  float rinv = 1.0f / lrow;
  _Float16* op = ctx + ((size_t)(b * Sc + st * 16 + lq)) * Ec + h * DHc;
#pragma unroll
  for (int t = 0; t < 4; ++t) {
    v8h pk;
#pragma unroll
    for (int j = 0; j < 8; ++j) pk[j] = (_Float16)(acc[t][j] * rinv);
    *(v8h*)(op + 16 * t + 8 * hs) = pk;
  }
}

// ---------------- output projection ----------------
__global__ void k_oproj(const _Float16* __restrict__ ctx, const _Float16* __restrict__ woh,
                        const float* __restrict__ bo, float* __restrict__ out) {
  const int tid  = blockIdx.x * 256 + threadIdx.x;
  const int wid  = tid >> 5;
  const int lane = threadIdx.x & 31;
  const int lq   = lane & 15;
  const int hs   = lane >> 4;

  const int ft = wid / (Bc * Sc / 16);
  const int st = wid % (Bc * Sc / 16);

  const _Float16* crow = ctx + (size_t)(st * 16 + lq) * Ec;
  v8f acc[4] = {};
  for (int e = 0; e < Ec; e += 32) {
    v16h bfr = load_bfrag(crow + e, hs);
#pragma unroll
    for (int t = 0; t < 4; ++t) {
      v16h afr = load_afrag(woh + (size_t)(ft * 64 + 16 * t + lq) * Ec + e, hs);
      acc[t] = wmma16(afr, bfr, acc[t]);
    }
  }
#pragma unroll
  for (int t = 0; t < 4; ++t) {
    const float* bp = bo + ft * 64 + 16 * t + 8 * hs;
    float* op = out + (size_t)(st * 16 + lq) * Ec + ft * 64 + 16 * t + 8 * hs;
    v4f r0, r1;
#pragma unroll
    for (int j = 0; j < 4; ++j) { r0[j] = acc[t][j] + bp[j]; r1[j] = acc[t][j + 4] + bp[j + 4]; }
    *(v4f*)op = r0;
    *(v4f*)(op + 4) = r1;
  }
}

// ---------------- host launch ----------------
extern "C" void kernel_launch(void* const* d_in, const int* in_sizes, int n_in,
                              void* d_out, int out_size, void* d_ws, size_t ws_size,
                              hipStream_t stream) {
  const float* x  = (const float*)d_in[0];
  const float* Wq = (const float*)d_in[1];
  const float* Wk = (const float*)d_in[2];
  const float* Wv = (const float*)d_in[3];
  const float* bq = (const float*)d_in[4];
  const float* bk = (const float*)d_in[5];
  const float* bv = (const float*)d_in[6];
  const float* Wo = (const float*)d_in[7];
  const float* bo = (const float*)d_in[8];
  float* out = (float*)d_out;

  const size_t BSE = (size_t)Bc * Sc * Ec;
  const size_t HDE = (size_t)Hc * DHc * Ec;
  const size_t EE  = (size_t)Ec * Ec;

  _Float16* ws  = (_Float16*)d_ws;
  _Float16* xh  = ws; ws += BSE;
  _Float16* wqt = ws; ws += HDE;
  _Float16* wkt = ws; ws += HDE;
  _Float16* wvt = ws; ws += HDE;
  _Float16* woh = ws; ws += EE;
  _Float16* qh  = ws; ws += BSE;
  _Float16* kh  = ws; ws += BSE;
  _Float16* vtp = ws; ws += BSE;
  _Float16* ctx = ws; ws += BSE;
  (void)ws_size; (void)in_sizes; (void)n_in; (void)out_size;

  k_cvt_f2h<<<(int)(BSE / 1024), 256, 0, stream>>>(x, xh, (int)BSE);
  k_cvt_wT <<<(int)(HDE / 256), 256, 0, stream>>>(Wq, wqt);
  k_cvt_wT <<<(int)(HDE / 256), 256, 0, stream>>>(Wk, wkt);
  k_cvt_wT <<<(int)(HDE / 256), 256, 0, stream>>>(Wv, wvt);
  k_cvt_f2h<<<(int)(EE / 1024), 256, 0, stream>>>(Wo, woh, (int)EE);

  k_qkv <<<3 * Bc * Hc * (Sc / 16) / 8, 256, 0, stream>>>(xh, wqt, wkt, wvt, bq, bk, bv,
                                                          qh, kh, vtp);
  k_attn<<<Bc * Hc * (Sc / 16) / 8, 256, 0, stream>>>(qh, kh, vtp, ctx);
  k_oproj<<<(Ec / 64) * (Bc * Sc / 16) / 8, 256, 0, stream>>>(ctx, woh, bo, out);
}
